// S4Encoder_63900523430526
// MI455X (gfx1250) — matrix operand order repaired and checked by reading, not compiled, not run on hardware
//
#include <hip/hip_runtime.h>

// ---------------------------------------------------------------------------
// S4 encoder for MI455X (gfx1250, wave32, WMMA + TDM).
//  - canonical activation layout: channels-last f32 [B*L, H]
//  - GEMMs: bf16 WMMA 16x16x32, f32 accumulate, fused epilogues
//  - weight panels staged into LDS by the Tensor Data Mover (one DMA / WG),
//    descriptor pad feature gives a conflict-free K+8 LDS row stride
//  - S4D long conv as diagonal complex recurrence (exact equivalent of the
//    reference FFT convolution), wave-per-(b,h), 2 complex states per lane
// ---------------------------------------------------------------------------

#define DEV __device__ __forceinline__

typedef __attribute__((ext_vector_type(16))) __bf16        bfx16;
typedef __attribute__((ext_vector_type(8)))  __bf16        bfx8;
typedef __attribute__((ext_vector_type(8)))  float         fx8;
typedef __attribute__((ext_vector_type(4)))  unsigned int  u32x4;
typedef __attribute__((ext_vector_type(8)))  int           i32x8;
typedef __attribute__((ext_vector_type(4)))  int           i32x4;

constexpr int Bn  = 16;
constexpr int Ln  = 4096;
constexpr int Hn  = 256;
constexpr int Nn  = 64;
constexpr int NBn = 4;
constexpr int Mn  = Bn * Ln;   // 65536 rows
constexpr int H2n = 2 * Hn;    // 512

DEV __bf16 f2bf(float f) {
  unsigned u = __float_as_uint(f);
  u += 0x7FFFu + ((u >> 16) & 1u);           // round-to-nearest-even
  unsigned short s = (unsigned short)(u >> 16);
  __bf16 r; __builtin_memcpy(&r, &s, 2); return r;
}

DEV float gelu_exact(float x) { return 0.5f * x * (1.0f + erff(x * 0.70710678118654752440f)); }
DEV float sigmoidf(float x)   { return 1.0f / (1.0f + expf(-x)); }

// ---------------- WMMA fragment helpers (contiguous k per lane half, same for A and B) ---------------

DEV bfx16 load_frag_a(const __bf16* __restrict__ A, int lda, int row, int k0, int hf) {
  // A: row-major [M, K].  lane<16: K {k0..k0+15}; lane>=16: K {k0+16..k0+31}.
  const __bf16* p = A + (long)row * lda + k0 + hf * 16;
  union { bfx16 v; bfx8 h[2]; } u;
  u.h[0] = *(const bfx8*)(p);
  u.h[1] = *(const bfx8*)(p + 8);
  return u.v;
}

DEV bfx16 load_frag_b_lds(const __bf16* bsm, int ldw, int rowN, int k0, int hf) {
  // LDS weight panel, row-major [rows, ldw].  lane holds column rowN,
  // 16 contiguous K values (lane<16: k0.., lane>=16: k0+16..).
  const __bf16* p = bsm + rowN * ldw + k0 + hf * 16;
  union { bfx16 v; bfx8 h[2]; } u;
  u.h[0] = *(const bfx8*)(p);
  u.h[1] = *(const bfx8*)(p + 8);
  return u.v;
}

DEV fx8 wmma_bf16(bfx16 a, bfx16 b, fx8 c) {
  return __builtin_amdgcn_wmma_f32_16x16x32_bf16(false, a, false, b, (short)0, c, false, false);
}

// -------------------- Tensor Data Mover (ISA ch.8) --------------------------
// 2-D tile DMA: global (row-major, row_elems bf16/row, stride = row_elems)
// -> LDS with 16B pad after each row (pad_interval picks the row length).
// This toolchain exposes the 6-arg builtin (extra int32x8 group, unused here).

DEV void tdm_load_panel(const __bf16* gsrc, unsigned lds_off, int rows,
                        int row_elems, int pad_interval_code) {
  unsigned long long ga = (unsigned long long)(uintptr_t)gsrc;
  u32x4 g0;
  g0.x = 1u;                                           // count=1, load, user D#
  g0.y = lds_off;                                      // LDS byte address
  g0.z = (unsigned)(ga & 0xFFFFFFFFu);                 // global addr [31:0]
  g0.w = (unsigned)((ga >> 32) & 0x1FFFFFFu)           // global addr [56:32]
         | 0x80000000u;                                // type = 2 ("image")
  unsigned td0 = (unsigned)row_elems;                  // dim0 len (elements)
  unsigned td1 = (unsigned)rows;                       // dim1 len (rows)
  i32x8 g1;
  g1[0] = (int)((1u << 16)                             // data_size = 2 bytes
                | (1u << 20)                           // pad_enable
                | ((unsigned)pad_interval_code << 22)  // DWORDs per row
                | (3u << 25));                         // pad_amount = 4 DWORDs
  g1[1] = (int)((td0 & 0xFFFFu) << 16);                // abar=0 | tensor_dim0 lo
  g1[2] = (int)(((td0 >> 16) & 0xFFFFu) | ((td1 & 0xFFFFu) << 16));
  g1[3] = (int)(((td1 >> 16) & 0xFFFFu) | (td0 << 16));// tile_dim0 = row_elems
  g1[4] = (int)(td1 & 0xFFFFu);                        // tile_dim1 | tile_dim2=0
  g1[5] = (int)td0;                                    // tensor_dim0_stride lo
  g1[6] = 0;                                           // stride0 hi | stride1 lo
  g1[7] = 0;                                           // stride1 hi
  i32x4 gz4 = {0, 0, 0, 0};                            // 2-D: groups 2/3 unused
  i32x8 gz8 = {0, 0, 0, 0, 0, 0, 0, 0};
  __builtin_amdgcn_tensor_load_to_lds(g0, g1, gz4, gz4, gz8, 0);
}

// ---------------------------- small kernels --------------------------------

__global__ void convert_weights_kernel(const float* __restrict__ s4w,
                                       const float* __restrict__ f1,
                                       const float* __restrict__ f2,
                                       __bf16* __restrict__ dst) {
  const int SEG = NBn * H2n * Hn;
  int id = blockIdx.x * 256 + threadIdx.x;      // < 3*SEG
  float v;
  if (id < SEG)           v = s4w[id];
  else if (id < 2 * SEG)  v = f1[id - SEG];
  else                    v = f2[id - 2 * SEG];
  dst[id] = f2bf(v);
}

__global__ void ssm_setup_kernel(const float* __restrict__ log_dt,
                                 const float* __restrict__ laR,
                                 const float* __restrict__ Aim,
                                 const float* __restrict__ Cre,
                                 const float* __restrict__ Cim,
                                 float4* __restrict__ sc) {
  int id = blockIdx.x * 256 + threadIdx.x;      // < NB*H*N
  int i  = id / (Hn * Nn);
  int hh = (id / Nn) % Hn;
  float dt = expf(log_dt[i * Hn + hh]);
  float Ar = -expf(laR[id]);
  float Ai = Aim[id];
  float m  = expf(dt * Ar);
  float sn, cs; sincosf(dt * Ai, &sn, &cs);
  float ar = m * cs, ai = m * sn;               // a = exp(dt*A)
  float e1r = ar - 1.0f, e1i = ai;              // exp(dt*A) - 1
  float inv = 1.0f / (Ar * Ar + Ai * Ai);
  float tr = (e1r * Ar + e1i * Ai) * inv;       // (e-1)/A
  float ti = (e1i * Ar - e1r * Ai) * inv;
  float cr = Cre[id] * tr - Cim[id] * ti;       // C' = C*(e-1)/A
  float ci = Cre[id] * ti + Cim[id] * tr;
  sc[id] = make_float4(ar, ai, cr, ci);
}

__global__ void conv_stem_kernel(const float* __restrict__ x,
                                 const float* __restrict__ w,
                                 const float* __restrict__ bias,
                                 float* __restrict__ h) {
  int id = blockIdx.x * 256 + threadIdx.x;      // < M*H, channel fastest
  int c = id & (Hn - 1);
  int m = id >> 8;
  int b = m >> 12;
  int l = m & (Ln - 1);
  float acc = bias[c];
#pragma unroll
  for (int t = 0; t < 5; ++t) {
    int li = l + t - 2;
    if (li >= 0 && li < Ln) acc = fmaf(w[c * 5 + t], x[b * Ln + li], acc);
  }
  h[id] = acc;
}

template <bool BF16OUT>
__global__ __launch_bounds__(256) void ln_kernel(const float* __restrict__ hin,
                                                 const float* __restrict__ w,
                                                 const float* __restrict__ bias,
                                                 void* __restrict__ outv) {
  int lane = threadIdx.x & 31;
  int wv   = threadIdx.x >> 5;
  long row = (long)blockIdx.x * 8 + wv;
  const float* p = hin + row * Hn + lane * 8;
  float xv[8];
#pragma unroll
  for (int j = 0; j < 8; ++j) xv[j] = p[j];
  float s = 0.f, ss = 0.f;
#pragma unroll
  for (int j = 0; j < 8; ++j) { s += xv[j]; ss = fmaf(xv[j], xv[j], ss); }
#pragma unroll
  for (int m = 16; m >= 1; m >>= 1) {
    s  += __shfl_xor(s,  m, 32);
    ss += __shfl_xor(ss, m, 32);
  }
  float mean = s * (1.0f / Hn);
  float var  = ss * (1.0f / Hn) - mean * mean;
  float rstd = rsqrtf(var + 1e-5f);
#pragma unroll
  for (int j = 0; j < 8; ++j) {
    int c = lane * 8 + j;
    float v = (xv[j] - mean) * rstd * w[c] + bias[c];
    if (BF16OUT) ((__bf16*)outv)[row * Hn + c] = f2bf(v);
    else         ((float*)outv)[row * Hn + c] = v;
  }
}

// S4D diagonal scan: wave per (b, h); lane holds states n=lane and n=lane+32.
// Fuses D-skip and exact GELU; writes bf16 GEMM operand.
__global__ __launch_bounds__(256) void ssm_kernel(const float* __restrict__ z,
                                                  const float4* __restrict__ sc,
                                                  const float* __restrict__ Dv,
                                                  __bf16* __restrict__ yg) {
  __shared__ float pbuf[8][33 * 32];   // padded (stride 33) transpose buffer
  __shared__ float zsh[8][32];
  int lane = threadIdx.x & 31;
  int wv   = threadIdx.x >> 5;
  int wid  = blockIdx.x * 8 + wv;      // 0 .. B*H-1
  int b    = wid >> 8;
  int hh   = wid & 255;
  float4 q0 = sc[hh * Nn + lane];        // a0, C'0
  float4 q1 = sc[hh * Nn + lane + 32];   // a1, C'1
  float Dh = Dv[hh];
  float s0r = 0.f, s0i = 0.f, s1r = 0.f, s1i = 0.f;
  const float* zrow = z  + (long)b * Ln * Hn + hh;
  __bf16*      yrow = yg + (long)b * Ln * Hn + hh;
  float* pw = pbuf[wv];
  float* zw = zsh[wv];
  for (int l0 = 0; l0 < Ln; l0 += 32) {
    float zc = zrow[(long)(l0 + lane) * Hn];
    zw[lane] = zc;
#pragma unroll
    for (int t = 0; t < 32; ++t) {
      float zt = zw[t];                               // broadcast step input
      float n0r = fmaf(q0.x, s0r, fmaf(-q0.y, s0i, zt));
      float n0i = fmaf(q0.x, s0i, q0.y * s0r);
      float n1r = fmaf(q1.x, s1r, fmaf(-q1.y, s1i, zt));
      float n1i = fmaf(q1.x, s1i, q1.y * s1r);
      s0r = n0r; s0i = n0i; s1r = n1r; s1i = n1i;
      float pp = q0.z * s0r - q0.w * s0i + q1.z * s1r - q1.w * s1i;
      pw[t * 33 + lane] = pp;
    }
    float acc = 0.f;
#pragma unroll
    for (int j = 0; j < 32; ++j) acc += pw[lane * 33 + j];   // lane = timestep
    float u = 2.0f * acc + zc * Dh;                          // conv + D-skip
    yrow[(long)(l0 + lane) * Hn] = f2bf(gelu_exact(u));
  }
}

// ------------------------------ GEMM kernels -------------------------------
// Block tile 128x64, 8 waves (4x2), wave tile 32x32 (2x2 WMMA frags).
// Weight panel (64 rows x K bf16) staged into LDS by one TDM DMA / WG.

enum { EPI_GELU_BF16 = 0, EPI_ADD_F32 = 1 };

template <int EPI, int K, int NOUT>
__global__ __launch_bounds__(256) void gemm_epi(const __bf16* __restrict__ A,
                                                const __bf16* __restrict__ W,
                                                const float* __restrict__ bias,
                                                void* __restrict__ out) {
  constexpr int LDSK = K + 8;                 // +16B pad/row (TDM pad feature)
  __shared__ __bf16 bsm[64 * LDSK];
  if (threadIdx.x < 32) {                     // wave 0 issues the panel DMA
    tdm_load_panel(W + (long)blockIdx.y * 64 * K,
                   (unsigned)(uintptr_t)(&bsm[0]), 64, K, (K == 256) ? 6 : 7);
    __builtin_amdgcn_s_wait_tensorcnt(0);
  }
  __syncthreads();

  int lane = threadIdx.x & 31;
  int wv   = threadIdx.x >> 5;
  int r    = lane & 15;
  int hf   = lane >> 4;
  int m0 = blockIdx.x * 128 + (wv >> 1) * 32;
  int nl = (wv & 1) * 32;                     // panel-local column base
  fx8 acc[2][2] = {};
  for (int k0 = 0; k0 < K; k0 += 32) {
    if (k0 + 32 < K) __builtin_prefetch(A + (long)(m0 + r) * K + k0 + 32, 0, 1);
    bfx16 a0 = load_frag_a(A, K, m0 + r,      k0, hf);
    bfx16 a1 = load_frag_a(A, K, m0 + 16 + r, k0, hf);
    bfx16 b0 = load_frag_b_lds(bsm, LDSK, nl + r,      k0, hf);
    bfx16 b1 = load_frag_b_lds(bsm, LDSK, nl + 16 + r, k0, hf);
    acc[0][0] = wmma_bf16(a0, b0, acc[0][0]);
    acc[0][1] = wmma_bf16(a0, b1, acc[0][1]);
    acc[1][0] = wmma_bf16(a1, b0, acc[1][0]);
    acc[1][1] = wmma_bf16(a1, b1, acc[1][1]);
  }
  float*  outf = (float*)out;
  __bf16* outb = (__bf16*)out;
#pragma unroll
  for (int tm = 0; tm < 2; ++tm)
#pragma unroll
    for (int tn = 0; tn < 2; ++tn) {
      int col  = blockIdx.y * 64 + nl + tn * 16 + r;
      int rowb = m0 + tm * 16 + hf * 8;
      float bv = bias[col];
#pragma unroll
      for (int v = 0; v < 8; ++v) {
        long idx = (long)(rowb + v) * NOUT + col;
        float a = acc[tm][tn][v] + bv;
        if (EPI == EPI_GELU_BF16) outb[idx] = f2bf(gelu_exact(a));
        else                      outf[idx] += a;
      }
    }
}

// GLU GEMM: per output column n computes both halves (rows n and n+256 of W),
// epilogue h += a * sigmoid(g).  K = H = 256, NOUT = H.
__global__ __launch_bounds__(256) void gemm_glu(const __bf16* __restrict__ A,
                                                const __bf16* __restrict__ W,
                                                const float* __restrict__ bias,
                                                float* __restrict__ h) {
  constexpr int K = Hn;
  constexpr int LDSK = K + 8;
  __shared__ __bf16 bsm[128 * LDSK];          // [0..63]=a-half, [64..127]=g-half
  if (threadIdx.x < 32) {
    unsigned base = (unsigned)(uintptr_t)(&bsm[0]);
    tdm_load_panel(W + (long)blockIdx.y * 64 * K,        base,                   64, K, 6);
    tdm_load_panel(W + (long)(Hn + blockIdx.y * 64) * K, base + 64 * LDSK * 2,  64, K, 6);
    __builtin_amdgcn_s_wait_tensorcnt(0);
  }
  __syncthreads();

  int lane = threadIdx.x & 31;
  int wv   = threadIdx.x >> 5;
  int r    = lane & 15;
  int hf   = lane >> 4;
  int m0 = blockIdx.x * 128 + (wv >> 1) * 32;
  int nl = (wv & 1) * 32;
  fx8 ca[2][2] = {};
  fx8 cg[2][2] = {};
  for (int k0 = 0; k0 < K; k0 += 32) {
    if (k0 + 32 < K) __builtin_prefetch(A + (long)(m0 + r) * K + k0 + 32, 0, 1);
    bfx16 a0  = load_frag_a(A, K, m0 + r,      k0, hf);
    bfx16 a1  = load_frag_a(A, K, m0 + 16 + r, k0, hf);
    bfx16 bA0 = load_frag_b_lds(bsm, LDSK, nl + r,            k0, hf);
    bfx16 bA1 = load_frag_b_lds(bsm, LDSK, nl + 16 + r,       k0, hf);
    bfx16 bG0 = load_frag_b_lds(bsm, LDSK, 64 + nl + r,       k0, hf);
    bfx16 bG1 = load_frag_b_lds(bsm, LDSK, 64 + nl + 16 + r,  k0, hf);
    ca[0][0] = wmma_bf16(a0, bA0, ca[0][0]);
    ca[0][1] = wmma_bf16(a0, bA1, ca[0][1]);
    ca[1][0] = wmma_bf16(a1, bA0, ca[1][0]);
    ca[1][1] = wmma_bf16(a1, bA1, ca[1][1]);
    cg[0][0] = wmma_bf16(a0, bG0, cg[0][0]);
    cg[0][1] = wmma_bf16(a0, bG1, cg[0][1]);
    cg[1][0] = wmma_bf16(a1, bG0, cg[1][0]);
    cg[1][1] = wmma_bf16(a1, bG1, cg[1][1]);
  }
#pragma unroll
  for (int tm = 0; tm < 2; ++tm)
#pragma unroll
    for (int tn = 0; tn < 2; ++tn) {
      int col  = blockIdx.y * 64 + nl + tn * 16 + r;
      int rowb = m0 + tm * 16 + hf * 8;
      float bA = bias[col];
      float bG = bias[col + Hn];
#pragma unroll
      for (int v = 0; v < 8; ++v) {
        float av = ca[tm][tn][v] + bA;
        float gv = cg[tm][tn][v] + bG;
        h[(long)(rowb + v) * Hn + col] += av * sigmoidf(gv);
      }
    }
}

__global__ void gather_out_kernel(const float* __restrict__ h, float* __restrict__ out) {
  int id = blockIdx.x * 256 + threadIdx.x;   // < B*H
  int b = id >> 8, c = id & 255;
  out[id] = h[(long)(b * Ln + Ln - 1) * Hn + c];
}

// ------------------------------- launcher ----------------------------------

extern "C" void kernel_launch(void* const* d_in, const int* in_sizes, int n_in,
                              void* d_out, int out_size, void* d_ws, size_t ws_size,
                              hipStream_t stream) {
  (void)in_sizes; (void)n_in; (void)out_size; (void)ws_size;
  const float* x          = (const float*)d_in[0];
  const float* conv_w     = (const float*)d_in[1];
  const float* conv_b     = (const float*)d_in[2];
  const float* s4_nw      = (const float*)d_in[3];
  const float* s4_nb      = (const float*)d_in[4];
  const float* log_dt     = (const float*)d_in[5];
  const float* log_A_real = (const float*)d_in[6];
  const float* A_imag     = (const float*)d_in[7];
  const float* C_re       = (const float*)d_in[8];
  const float* C_im       = (const float*)d_in[9];
  const float* Dp         = (const float*)d_in[10];
  const float* s4_out_w   = (const float*)d_in[11];
  const float* s4_out_b   = (const float*)d_in[12];
  const float* ff_nw      = (const float*)d_in[13];
  const float* ff_nb      = (const float*)d_in[14];
  const float* ff_w1      = (const float*)d_in[15];
  const float* ff_b1      = (const float*)d_in[16];
  const float* ff_w2      = (const float*)d_in[17];
  const float* ff_b2      = (const float*)d_in[18];
  float* out = (float*)d_out;

  // Workspace layout (all regions 256B aligned):
  //   h     f32 [M,H]                        64 MB
  //   zg    f32 z [M,H] / bf16 g [M,2H]      64 MB (phase-disjoint share)
  //   act   bf16 [M,H] (yg / zb)             32 MB
  //   wts   bf16 weights                      3 MB
  //   sc    f32x4 [NB,H,N]                    1 MB
  char* wsp = (char*)d_ws;
  float*  hbuf = (float*)wsp;
  char*   zgp  = wsp + (size_t)Mn * Hn * 4;
  float*  zbuf = (float*)zgp;
  __bf16* gbuf = (__bf16*)zgp;
  __bf16* act  = (__bf16*)(zgp + (size_t)Mn * H2n * 2);
  __bf16* wts  = (__bf16*)((char*)act + (size_t)Mn * Hn * 2);
  const int SEG = NBn * H2n * Hn;          // 524288 weights per family
  __bf16* w_s4 = wts;
  __bf16* w_f1 = wts + SEG;
  __bf16* w_f2 = wts + 2 * SEG;
  float4* sc   = (float4*)((char*)wts + (size_t)3 * SEG * 2);

  convert_weights_kernel<<<3 * SEG / 256, 256, 0, stream>>>(s4_out_w, ff_w1, ff_w2, wts);
  ssm_setup_kernel<<<NBn * Hn * Nn / 256, 256, 0, stream>>>(log_dt, log_A_real, A_imag,
                                                            C_re, C_im, sc);
  conv_stem_kernel<<<(Mn * Hn) / 256, 256, 0, stream>>>(x, conv_w, conv_b, hbuf);

  for (int i = 0; i < NBn; ++i) {
    // --- S4 residual block ---
    ln_kernel<false><<<Mn / 8, 256, 0, stream>>>(hbuf, s4_nw + i * Hn, s4_nb + i * Hn, zbuf);
    ssm_kernel<<<(Bn * Hn) / 8, 256, 0, stream>>>(zbuf, sc + (size_t)i * Hn * Nn,
                                                  Dp + i * Hn, act);
    gemm_glu<<<dim3(Mn / 128, Hn / 64), 256, 0, stream>>>(act, w_s4 + (size_t)i * H2n * Hn,
                                                          s4_out_b + i * H2n, hbuf);
    // --- FF residual block ---
    ln_kernel<true><<<Mn / 8, 256, 0, stream>>>(hbuf, ff_nw + i * Hn, ff_nb + i * Hn, act);
    gemm_epi<EPI_GELU_BF16, Hn, H2n><<<dim3(Mn / 128, H2n / 64), 256, 0, stream>>>(
        act, w_f1 + (size_t)i * H2n * Hn, ff_b1 + i * H2n, gbuf);
    gemm_epi<EPI_ADD_F32, H2n, Hn><<<dim3(Mn / 128, Hn / 64), 256, 0, stream>>>(
        gbuf, w_f2 + (size_t)i * Hn * H2n, ff_b2 + i * Hn, hbuf);
  }

  gather_out_kernel<<<(Bn * Hn) / 256, 256, 0, stream>>>(hbuf, out);
}
